// SpanNNShot_30442728194236
// MI455X (gfx1250) — compile-verified, matrix operand order checked by reading, not run on hardware
//
#include <hip/hip_runtime.h>

// ---------------------------------------------------------------------------
// SpanNNShot fused pipeline for MI455X (gfx1250, wave32, WMMA + TDM)
//   T=32768, D=768, NS=NQ=8192, L=6
//   dist = 2*Q@S^T - |Q|^2 - |S|^2 ; logits = per-label max ; pred = argmax
// ---------------------------------------------------------------------------

#define T_TOK   32768
#define D_TOK   768
#define NS      8192
#define NQ      8192
#define NL      6
#define KDIM    1536          // 2*D

#define BM      128           // query rows per workgroup
#define BNSUB   128           // support cols per K-sweep
#define CHUNK   1024          // support cols per workgroup
#define LDA     40            // padded LDS row stride in bf16 elems (80B, 16B aligned)
#define NK      (KDIM / 32)   // 48 K-steps

typedef __attribute__((ext_vector_type(16))) __bf16 bf16x16;
typedef __attribute__((ext_vector_type(8)))  __bf16 bf16x8;
typedef __attribute__((ext_vector_type(8)))  float  v8f;
typedef __attribute__((ext_vector_type(4)))  unsigned int u32x4;
typedef __attribute__((ext_vector_type(8)))  int  i32x8;
typedef __attribute__((ext_vector_type(4)))  int  i32x4;

#define HAS_TDM (__has_builtin(__builtin_amdgcn_tensor_load_to_lds) && \
                 __has_builtin(__builtin_amdgcn_s_wait_tensorcnt))

// ---- order-preserving float <-> uint encoding (for atomic max) ------------
__device__ __forceinline__ unsigned encf(float f) {
    unsigned u = __float_as_uint(f);
    return (u & 0x80000000u) ? ~u : (u | 0x80000000u);
}
__device__ __forceinline__ float decf(unsigned u) {
    unsigned b = (u & 0x80000000u) ? (u & 0x7FFFFFFFu) : ~u;
    return __uint_as_float(b);
}
#define ENC_NEG_FLT_MAX 0x00800000u   // encf(-FLT_MAX)

// fp32 -> bf16 (round to nearest even)
__device__ __forceinline__ unsigned short f2bf(float f) {
    unsigned u = __float_as_uint(f);
    unsigned r = (u + 0x7FFFu + ((u >> 16) & 1u)) >> 16;
    return (unsigned short)r;
}

// load one 16x32 bf16 WMMA fragment row-slice for this lane from LDS
__device__ __forceinline__ bf16x16 ld_frag(const __bf16* p) {
    bf16x8 lo = *(const bf16x8*)(p);
    bf16x8 hi = *(const bf16x8*)(p + 16);
    bf16x16 r;
#pragma unroll
    for (int i = 0; i < 8; ++i) { r[i] = lo[i]; r[i + 8] = hi[i]; }
    return r;
}

#if HAS_TDM
// ---------------------------------------------------------------------------
// TDM: DMA a 128-row x 32-col bf16 tile from a row-major [8192 x 1536] tensor
// into LDS at lds_addr, inserting 16B of padding after every 64B of row data
// (-> 80B LDS row stride == LDA*2). Tracked by TENSORcnt. D# layout per
// CDNA5 ISA (groups 0/1; 2D so groups 2/3 are zero).
// 6-arg builtin form (amdgpu-toolchain / clang-23): trailing i32x8 is the
// unused extra descriptor payload, then the cache-policy immediate.
// ---------------------------------------------------------------------------
__device__ __forceinline__ void tdm_load_tile(unsigned lds_addr,
                                              unsigned long long gaddr) {
    u32x4 g0;
    g0[0] = 1u;                                              // count=1, user mode
    g0[1] = lds_addr;                                        // LDS byte address
    g0[2] = (unsigned)(gaddr & 0xFFFFFFFFull);               // global_addr[31:0]
    g0[3] = (unsigned)((gaddr >> 32) & 0x01FFFFFFull)        // global_addr[56:32]
            | (2u << 30);                                    // type = 2 ("image")
    i32x8 g1;
    g1[0] = (int)((1u << 16)      // data_size = 1 -> 2 bytes
                | (1u << 20)      // pad_enable
                | (3u << 22)      // pad_interval = 3 -> every 16 DWORDs (64B)
                | (3u << 25));    // pad_amount   = 3 -> 4 DWORDs (16B)
    g1[1] = (int)(((unsigned)KDIM & 0xFFFFu) << 16);         // tensor_dim0[15:0]=1536
    g1[2] = (int)(((unsigned)NS & 0xFFFFu) << 16);           // tensor_dim1[15:0]=8192
    g1[3] = (int)(32u << 16);                                // tile_dim0 = 32
    g1[4] = 128;                                             // tile_dim1 = 128
    g1[5] = KDIM;                                            // tensor_dim0_stride
    g1[6] = 0;
    g1[7] = 0;
    i32x4 z4 = {0, 0, 0, 0};
    i32x8 z8 = {0, 0, 0, 0, 0, 0, 0, 0};
    __builtin_amdgcn_tensor_load_to_lds(g0, g1, z4, z4, z8, 0);
}
#endif

// ---------------------------------------------------------------------------
// Stage 0a: histogram labels, write per-label base offsets (cursors)
// ---------------------------------------------------------------------------
__global__ void k_count(const int* __restrict__ labels, int* __restrict__ cursor) {
    __shared__ int c[8];
    if (threadIdx.x < 8) c[threadIdx.x] = 0;
    __syncthreads();
    for (int s = threadIdx.x; s < NS; s += blockDim.x)
        atomicAdd(&c[labels[s]], 1);
    __syncthreads();
    if (threadIdx.x == 0) {
        int run = 0;
        for (int l = 0; l < NL; ++l) { cursor[l] = run; run += c[l]; }
    }
}

// Stage 0b: scatter supports into label-contiguous order
__global__ void k_scatter(const int* __restrict__ labels, int* __restrict__ cursor,
                          int* __restrict__ perm, int* __restrict__ labp) {
    int s = blockIdx.x * blockDim.x + threadIdx.x;
    if (s < NS) {
        int lab = labels[s];
        int pos = atomicAdd(&cursor[lab], 1);
        perm[pos] = s;
        labp[pos] = lab;
    }
}

// ---------------------------------------------------------------------------
// Stage 1: build span embeddings (bf16) + fp32 squared norms
// ---------------------------------------------------------------------------
__global__ __launch_bounds__(256)
void k_embed(const float* __restrict__ tok, const int* __restrict__ st,
             const int* __restrict__ en, const int* __restrict__ perm,
             unsigned short* __restrict__ out, float* __restrict__ nrm) {
    int row = blockIdx.x;
    int j = perm ? perm[row] : row;
    long a = (long)st[j] * D_TOK;
    long b = (long)en[j] * D_TOK;
    unsigned short* orow = out + (size_t)row * KDIM;
    float part = 0.f;
#pragma unroll
    for (int i = 0; i < 6; ++i) {
        int e = threadIdx.x + i * 256;
        float v = (e < D_TOK) ? tok[a + e] : tok[b + (e - D_TOK)];
        part += v * v;
        orow[e] = f2bf(v);
    }
    __shared__ float red[256];
    red[threadIdx.x] = part;
    __syncthreads();
    for (int s = 128; s > 0; s >>= 1) {
        if (threadIdx.x < s) red[threadIdx.x] += red[threadIdx.x + s];
        __syncthreads();
    }
    if (threadIdx.x == 0) nrm[row] = red[0];
}

// Stage 1b: init global per-(q,label) max accumulator to encf(-FLT_MAX)
__global__ void k_init(unsigned* __restrict__ glmax) {
    int i = blockIdx.x * blockDim.x + threadIdx.x;
    if (i < NQ * NL) glmax[i] = ENC_NEG_FLT_MAX;
}

// ---------------------------------------------------------------------------
// Stage 2: fused bf16 WMMA GEMM (2*Q@S^T) + per-label running max epilogue
// grid: (NQ/BM, NS/CHUNK), 256 threads = 8 waves (4 M-waves x 2 N-waves)
// Double-buffered LDS tiles fed by the Tensor Data Mover (wave 0 issues,
// TENSORcnt + workgroup barrier publishes).
// ---------------------------------------------------------------------------
__global__ __launch_bounds__(256)
void k_gemm(const unsigned short* __restrict__ Qb, const unsigned short* __restrict__ Sb,
            const float* __restrict__ q2, const float* __restrict__ s2,
            const int* __restrict__ labp, unsigned* __restrict__ glmax) {
    __shared__ __bf16  lds_a[2][BM * LDA];
    __shared__ __bf16  lds_b[2][BNSUB * LDA];
    __shared__ unsigned wgmax[BM * NL];
    __shared__ float    q2l[BM];

    const int tid    = threadIdx.x;
    const int qrow0  = blockIdx.x * BM;
    const int chunk0 = blockIdx.y * CHUNK;

    if (tid < BM) q2l[tid] = q2[qrow0 + tid];
    for (int i = tid; i < BM * NL; i += 256) wgmax[i] = ENC_NEG_FLT_MAX;

    const int lane = tid & 31;
    const int wave = tid >> 5;
    const int wm   = wave & 3;    // 0..3 : 32-row strip
    const int wn   = wave >> 2;   // 0..1 : 64-col strip
    const int m    = lane & 15;
    const int kg   = lane >> 4;

#if HAS_TDM
    const unsigned ldsA[2] = { (unsigned)(unsigned long long)(size_t)&lds_a[0][0],
                               (unsigned)(unsigned long long)(size_t)&lds_a[1][0] };
    const unsigned ldsB[2] = { (unsigned)(unsigned long long)(size_t)&lds_b[0][0],
                               (unsigned)(unsigned long long)(size_t)&lds_b[1][0] };
    const unsigned long long gA0 =
        (unsigned long long)(size_t)Qb + (unsigned long long)qrow0 * KDIM * 2ull;
#else
    const int srow  = tid >> 1;
    const int shalf = tid & 1;
#endif

    __syncthreads();

    for (int nt = 0; nt < CHUNK / BNSUB; ++nt) {
        const int col0 = chunk0 + nt * BNSUB;
#if HAS_TDM
        const unsigned long long gB0 =
            (unsigned long long)(size_t)Sb + (unsigned long long)col0 * KDIM * 2ull;
#endif

        const v8f vzero = {0.f, 0.f, 0.f, 0.f, 0.f, 0.f, 0.f, 0.f};
        v8f acc[2][4];
#pragma unroll
        for (int mi = 0; mi < 2; ++mi)
#pragma unroll
            for (int ni = 0; ni < 4; ++ni) acc[mi][ni] = vzero;

#if HAS_TDM
        int buf = 0;
        if (tid < 32) {                       // wave 0 drives the TDM
            tdm_load_tile(ldsA[0], gA0);
            tdm_load_tile(ldsB[0], gB0);
        }
#endif
        for (int step = 0; step < NK; ++step) {
            const int kk = step * 32;
#if HAS_TDM
            if (tid < 32) {
                if (step + 1 < NK) {          // software pipeline: next slab in flight
                    tdm_load_tile(ldsA[buf ^ 1], gA0 + (unsigned long long)(kk + 32) * 2ull);
                    tdm_load_tile(ldsB[buf ^ 1], gB0 + (unsigned long long)(kk + 32) * 2ull);
                    __builtin_amdgcn_s_wait_tensorcnt(2);  // current pair landed
                } else {
                    __builtin_amdgcn_s_wait_tensorcnt(0);
                }
            }
            __syncthreads();                  // publish current buffer to all waves
            const __bf16* ta = &lds_a[buf][0];
            const __bf16* tb = &lds_b[buf][0];
#else
            __syncthreads();                  // previous iteration's reads done
            {
                const uint4* g = (const uint4*)(Qb + (size_t)(qrow0 + srow) * KDIM + kk + shalf * 16);
                uint4 d0 = g[0], d1 = g[1];
                uint4* l = (uint4*)(&lds_a[0][srow * LDA + shalf * 16]);
                l[0] = d0; l[1] = d1;
                __builtin_prefetch(g + 4, 0, 3);
            }
            {
                const uint4* g = (const uint4*)(Sb + (size_t)(col0 + srow) * KDIM + kk + shalf * 16);
                uint4 d0 = g[0], d1 = g[1];
                uint4* l = (uint4*)(&lds_b[0][srow * LDA + shalf * 16]);
                l[0] = d0; l[1] = d1;
                __builtin_prefetch(g + 4, 0, 3);
            }
            __syncthreads();
            const __bf16* ta = &lds_a[0][0];
            const __bf16* tb = &lds_b[0][0];
#endif
            bf16x16 af[2], bfr[4];
#pragma unroll
            for (int mi = 0; mi < 2; ++mi)
                af[mi] = ld_frag(&ta[(wm * 32 + mi * 16 + m) * LDA + kg * 8]);
#pragma unroll
            for (int ni = 0; ni < 4; ++ni)
                bfr[ni] = ld_frag(&tb[(wn * 64 + ni * 16 + m) * LDA + kg * 8]);

#pragma unroll
            for (int mi = 0; mi < 2; ++mi)
#pragma unroll
                for (int ni = 0; ni < 4; ++ni)
                    acc[mi][ni] = __builtin_amdgcn_wmma_f32_16x16x32_bf16(
                        false, af[mi], false, bfr[ni], (short)0, acc[mi][ni], false, false);

            __syncthreads();                  // reads done; buffer may be refilled
#if HAS_TDM
            buf ^= 1;
#endif
        }

        // epilogue: dist = 2*acc - |q|^2 - |s|^2 ; per-label LDS atomic max
#pragma unroll
        for (int ni = 0; ni < 4; ++ni) {
            const int c     = col0 + wn * 64 + ni * 16 + m;   // label-grouped support idx
            const float s2v = s2[c];
            const int lab   = labp[c];
#pragma unroll
            for (int mi = 0; mi < 2; ++mi) {
                const int rbase = wm * 32 + mi * 16 + kg * 8;
#pragma unroll
                for (int r = 0; r < 8; ++r) {
                    float d = 2.0f * acc[mi][ni][r] - q2l[rbase + r] - s2v;
                    atomicMax(&wgmax[(rbase + r) * NL + lab], encf(d));
                }
            }
        }
    }

    __syncthreads();
    for (int i = tid; i < BM * NL; i += 256)
        atomicMax(&glmax[(size_t)(qrow0 + i / NL) * NL + (i % NL)], wgmax[i]);
}

// ---------------------------------------------------------------------------
// Stage 3: decode logits, argmax -> pred
// ---------------------------------------------------------------------------
__global__ void k_final(const unsigned* __restrict__ glmax, float* __restrict__ out) {
    int q = blockIdx.x * blockDim.x + threadIdx.x;
    if (q >= NQ) return;
    float bv = 0.f;
    int bi = 0;
#pragma unroll
    for (int l = 0; l < NL; ++l) {
        float v = decf(glmax[q * NL + l]);
        out[q * NL + l] = v;
        if (l == 0 || v > bv) { bv = v; bi = l; }
    }
    ((int*)out)[(size_t)NQ * NL + q] = bi;   // pred (int32), flat after logits
}

// ---------------------------------------------------------------------------
extern "C" void kernel_launch(void* const* d_in, const int* in_sizes, int n_in,
                              void* d_out, int out_size, void* d_ws, size_t ws_size,
                              hipStream_t stream) {
    const float* tok_s   = (const float*)d_in[0];
    const float* tok_q   = (const float*)d_in[1];
    const int*   start_s = (const int*)d_in[2];
    const int*   end_s   = (const int*)d_in[3];
    const int*   start_q = (const int*)d_in[4];
    const int*   end_q   = (const int*)d_in[5];
    const int*   labels  = (const int*)d_in[6];
    // d_in[7] = num_labels (fixed at 6 for this problem)

    uint8_t* w = (uint8_t*)d_ws;
    size_t off = 0;
    auto take = [&](size_t bytes) { void* p = w + off; off = (off + bytes + 255) & ~(size_t)255; return p; };

    int*            cursor = (int*)take(64);
    int*            perm   = (int*)take((size_t)NS * 4);
    int*            labp   = (int*)take((size_t)NS * 4);
    float*          q2     = (float*)take((size_t)NQ * 4);
    float*          s2     = (float*)take((size_t)NS * 4);
    unsigned*       glmax  = (unsigned*)take((size_t)NQ * NL * 4);
    unsigned short* Sb     = (unsigned short*)take((size_t)NS * KDIM * 2);
    unsigned short* Qb     = (unsigned short*)take((size_t)NQ * KDIM * 2);

    k_count  <<<1, 256, 0, stream>>>(labels, cursor);
    k_scatter<<<NS / 256, 256, 0, stream>>>(labels, cursor, perm, labp);
    k_embed  <<<NS, 256, 0, stream>>>(tok_s, start_s, end_s, perm, Sb, s2);
    k_embed  <<<NQ, 256, 0, stream>>>(tok_q, start_q, end_q, nullptr, Qb, q2);
    k_init   <<<(NQ * NL) / 256, 256, 0, stream>>>(glmax);
    k_gemm   <<<dim3(NQ / BM, NS / CHUNK), 256, 0, stream>>>(Qb, Sb, q2, s2, labp, glmax);
    k_final  <<<NQ / 256, 256, 0, stream>>>(glmax, (float*)d_out);
}
